// GPTLanguageModel_83030307766304
// MI455X (gfx1250) — matrix operand order repaired and checked by reading, not compiled, not run on hardware
//
#include <hip/hip_runtime.h>
#include <hip/hip_bf16.h>
#include <math.h>

// ---------------------------------------------------------------------------
// Model constants (GPT: V=50257, D=512, FF=2048, H=8, HD=64, L=4, B=2, S=2048)
// ---------------------------------------------------------------------------
#define GPT_V  50257
#define GPT_D  512
#define GPT_FF 2048
#define GPT_L  4
#define GPT_B  2
#define GPT_S  2048
#define GPT_H  8
#define GPT_HD 64
#define GPT_N  (GPT_B * GPT_S)   // 4096 tokens

typedef __attribute__((ext_vector_type(16))) __bf16 bf16x16;
typedef __attribute__((ext_vector_type(8)))  __bf16 bf16x8;
typedef __attribute__((ext_vector_type(8)))  float  v8f;
typedef __attribute__((ext_vector_type(4)))  unsigned int u32x4;
typedef __attribute__((ext_vector_type(8)))  int  i32x8;
typedef __attribute__((ext_vector_type(4)))  int  i32x4;

static __device__ __forceinline__ v8f vzero8() {
    v8f c;
#pragma unroll
    for (int i = 0; i < 8; ++i) c[i] = 0.0f;
    return c;
}

static __device__ __forceinline__ v8f wmma_bf16(bf16x16 a, bf16x16 b, v8f c) {
    // D = A(16x32 bf16) * B(32x16 bf16) + C(16x16 f32)
    return __builtin_amdgcn_wmma_f32_16x16x32_bf16(
        /*neg_a=*/false, a, /*neg_b=*/false, b,
        /*c_mod=*/(short)0, c, /*reuse_a=*/false, /*reuse_b=*/false);
}

// A fragment: 16x32 bf16 tile at (row tile base, k0), row-major lda (elements).
// Contiguous k order used here (not the ISA table order): lane row = lane&15; lanes 0-15 hold K = {0..15},
// lanes 16-31 hold K = {16..31}.
static __device__ __forceinline__ bf16x16 load_a_frag(const __bf16* base, int lda, int k0) {
    int lane = threadIdx.x & 31;
    int row = lane & 15;
    int hs  = lane >> 4;
    const __bf16* p = base + (size_t)row * lda + k0 + hs * 16;
    union { bf16x16 v; bf16x8 h[2]; } u;
    u.h[0] = *(const bf16x8*)(p);
    u.h[1] = *(const bf16x8*)(p + 8);
    return u.v;
}

// ---------------------------------------------------------------------------
// Elementwise prep kernels
// ---------------------------------------------------------------------------
__global__ void convert_f32_bf16_kernel(const float* __restrict__ in,
                                        __bf16* __restrict__ out, int total) {
    int g = blockIdx.x * blockDim.x + threadIdx.x;
    if (g < total) out[g] = (__bf16)in[g];
}

// in: [R,C] f32 row-major  ->  out: [C,R] bf16 row-major (i.e. transpose)
__global__ void transpose_f32_bf16_kernel(const float* __restrict__ in,
                                          __bf16* __restrict__ out,
                                          int R, int C, int total) {
    int g = blockIdx.x * blockDim.x + threadIdx.x;
    if (g >= total) return;
    int r = g / C;
    int c = g - r * C;
    out[(size_t)c * R + r] = (__bf16)in[g];
}

// x[b,s,:] = wte[idx[b,s],:] + wpe[s,:]
__global__ void embed_kernel(const int* __restrict__ idx,
                             const float* __restrict__ wte,
                             const float* __restrict__ wpe,
                             float* __restrict__ x, int total) {
    int g = blockIdx.x * blockDim.x + threadIdx.x;
    if (g >= total) return;
    int d = g & (GPT_D - 1);
    int t = g >> 9;               // token index (D = 512 = 2^9)
    int s = t & (GPT_S - 1);      // position within sequence
    x[g] = wte[(size_t)idx[t] * GPT_D + d] + wpe[(size_t)s * GPT_D + d];
}

// LayerNorm over D=512, one 256-thread block per token row; bf16 output.
__global__ void ln_kernel(const float* __restrict__ x,
                          const float* __restrict__ w,
                          const float* __restrict__ b,
                          __bf16* __restrict__ out) {
    int row = blockIdx.x;
    int tid = threadIdx.x;       // 0..255
    const float* xr = x + (size_t)row * GPT_D;
    float v0 = xr[tid];
    float v1 = xr[tid + 256];
    __shared__ float sh[256];
    sh[tid] = v0 + v1;
    __syncthreads();
    for (int s = 128; s > 0; s >>= 1) {
        if (tid < s) sh[tid] += sh[tid + s];
        __syncthreads();
    }
    float mean = sh[0] * (1.0f / GPT_D);
    __syncthreads();
    float d0 = v0 - mean, d1 = v1 - mean;
    sh[tid] = d0 * d0 + d1 * d1;
    __syncthreads();
    for (int s = 128; s > 0; s >>= 1) {
        if (tid < s) sh[tid] += sh[tid + s];
        __syncthreads();
    }
    float rstd = rsqrtf(sh[0] * (1.0f / GPT_D) + 1e-5f);
    out[(size_t)row * GPT_D + tid]       = (__bf16)(d0 * rstd * w[tid] + b[tid]);
    out[(size_t)row * GPT_D + tid + 256] = (__bf16)(d1 * rstd * w[tid + 256] + b[tid + 256]);
}

// ---------------------------------------------------------------------------
// bf16 WMMA GEMM: C[M,N] = A[M,K] * Bt[N,K]^T (+bias), fused epilogue.
//
// Block = (32,4): 4 waves. The block's 16xK A strip is staged into LDS ONCE
// by the Tensor Data Mover (tensor_load_to_lds, waited with s_wait_tensorcnt),
// then each wave computes a 16x64 output slab (4 accumulators), reusing one
// A fragment (ds_load_b128 from LDS) across 4 B fragments per K-step.
// ---------------------------------------------------------------------------
#define MODE_PLAIN 0   // outf = val                          (logits)
#define MODE_RESID 1   // outf += val                         (proj / fc2 residual)
#define MODE_GELU  2   // outb = gelu(val)   bf16             (fc)
#define MODE_QKV   3   // scatter to Q(scaled)/K/[V transposed] bf16 head layout

__global__ void gemm_bf16_wmma(const __bf16* __restrict__ A,
                               const __bf16* __restrict__ Bt,
                               const float* __restrict__ bias,
                               float* __restrict__ outf,
                               __bf16* __restrict__ outb,
                               __bf16* __restrict__ qout,
                               __bf16* __restrict__ kout,
                               __bf16* __restrict__ vtout,
                               int M, int N, int K, int mode) {
    extern __shared__ __bf16 smemA[];          // [16][K] bf16
    int lane = threadIdx.x;
    int ln15 = lane & 15;
    int hs   = lane >> 4;
    int wave = threadIdx.y;                    // 0..3
    int m0   = blockIdx.y * 16;
    int n0   = blockIdx.x * 256 + wave * 64;   // wave's 64-wide N slab

    // ---- TDM: stage A[m0..m0+15][0..K) into LDS (one wave issues) ----
    if (wave == 0) {
        // Generic pointers to LDS carry the byte offset in their low 32 bits.
        unsigned lds_off = (unsigned)(size_t)(&smemA[0]);
        unsigned long long ga = (unsigned long long)(const void*)(A + (size_t)m0 * K);
        unsigned kd = (unsigned)K;
        u32x4 g0;
        g0[0] = 1u;                                            // count=1, user mode
        g0[1] = lds_off;                                       // lds_addr
        g0[2] = (unsigned)(ga & 0xffffffffu);                  // global_addr[31:0]
        g0[3] = (unsigned)((ga >> 32) & 0x01ffffffu) | (2u << 30); // addr[56:32] | type=2
        i32x8 g1;
        g1[0] = (int)(1u << 16);                               // data_size=1 (2 bytes)
        g1[1] = (int)((kd & 0xffffu) << 16);                   // tensor_dim0[15:0]
        g1[2] = (int)((kd >> 16) | (16u << 16));               // tensor_dim0[31:16] | tensor_dim1 lo
        g1[3] = (int)(kd << 16);                               // tensor_dim1 hi=0 | tile_dim0=K
        g1[4] = 16;                                            // tile_dim1=16 | tile_dim2=0
        g1[5] = (int)kd;                                       // tensor_dim0_stride[31:0]
        g1[6] = 0;                                             // stride hi | dim1_stride lo
        g1[7] = 0;
        i32x4 z4;
        z4[0] = 0; z4[1] = 0; z4[2] = 0; z4[3] = 0;
        i32x8 z8;
#pragma unroll
        for (int i = 0; i < 8; ++i) z8[i] = 0;
        __builtin_amdgcn_tensor_load_to_lds(g0, g1, z4, z4, z8, 0);
        __builtin_amdgcn_s_wait_tensorcnt(0);
    }
    __syncthreads();

    // ---- clamp ragged-edge columns for B loads (stores are guarded) ----
    const __bf16* bcol[4];
#pragma unroll
    for (int t = 0; t < 4; ++t) {
        int n = n0 + t * 16 + ln15;
        int ncl = (n < N) ? n : (N - 1);
        bcol[t] = Bt + (size_t)ncl * K;
    }

    v8f acc[4];
#pragma unroll
    for (int t = 0; t < 4; ++t) acc[t] = vzero8();

    for (int k0 = 0; k0 < K; k0 += 32) {
        union { bf16x16 v; bf16x8 h[2]; } ua;  // A fragment from LDS
        const __bf16* ap = smemA + (size_t)ln15 * K + k0 + hs * 16;
        ua.h[0] = *(const bf16x8*)(ap);
        ua.h[1] = *(const bf16x8*)(ap + 8);
#pragma unroll
        for (int t = 0; t < 4; ++t) {
            bf16x16 b = *(const bf16x16*)(bcol[t] + k0 + hs * 16);
            acc[t] = wmma_bf16(ua.v, b, acc[t]);
        }
    }

    // ---- fused epilogue ----
#pragma unroll
    for (int t = 0; t < 4; ++t) {
        int n = n0 + t * 16 + ln15;
        if (n >= N) continue;
        float bval = bias ? bias[n] : 0.0f;
#pragma unroll
        for (int i = 0; i < 8; ++i) {
            int m = m0 + i + 8 * hs;
            float val = acc[t][i] + bval;
            if (mode == MODE_PLAIN) {
                outf[(size_t)m * N + n] = val;
            } else if (mode == MODE_RESID) {
                outf[(size_t)m * N + n] += val;
            } else if (mode == MODE_GELU) {
                float g = 0.5f * val * (1.0f + erff(val * 0.70710678118654752f));
                outb[(size_t)m * N + n] = (__bf16)g;
            } else { // MODE_QKV: split heads; Q pre-scaled by 1/sqrt(HD)=0.125
                int bb = m >> 11;                  // S = 2048
                int ss = m & (GPT_S - 1);
                int hh = (n >> 6) & (GPT_H - 1);
                int dd = n & (GPT_HD - 1);
                if (n < GPT_D) {
                    qout[(((size_t)(bb * GPT_H + hh) * GPT_S + ss) << 6) + dd] =
                        (__bf16)(val * 0.125f);
                } else if (n < 2 * GPT_D) {
                    kout[(((size_t)(bb * GPT_H + hh) * GPT_S + ss) << 6) + dd] =
                        (__bf16)val;
                } else { // V stored transposed: [B,H,HD,S] for contiguous B-frags
                    vtout[((size_t)(bb * GPT_H + hh) * GPT_HD + dd) * GPT_S + ss] =
                        (__bf16)val;
                }
            }
        }
    }
}

// ---------------------------------------------------------------------------
// Flash attention: one wave per 16 query rows of one (b,h).
// Q pre-scaled; scores = Q(16x64) @ K^T via 2 WMMAs; online softmax with
// 16-lane shuffles; P bounced via 512B LDS (C-layout -> A-layout); O += P@V
// via 4 WMMAs against transposed V.
// ---------------------------------------------------------------------------
__global__ void flash_attn_kernel(const __bf16* __restrict__ Q,
                                  const __bf16* __restrict__ Kb,
                                  const __bf16* __restrict__ Vt,
                                  __bf16* __restrict__ O) {
    int bh  = blockIdx.x;        // 0..B*H-1
    int qt  = blockIdx.y;        // 0..S/16-1
    int lane = threadIdx.x;
    int ln15 = lane & 15;
    int hs   = lane >> 4;
    int q0 = qt * 16;

    const __bf16* qbase = Q + ((size_t)bh * GPT_S + q0) * GPT_HD;
    bf16x16 aq0 = load_a_frag(qbase, GPT_HD, 0);
    bf16x16 aq1 = load_a_frag(qbase, GPT_HD, 32);

    float mrow[8], lrow[8];
    v8f oacc[4];
#pragma unroll
    for (int i = 0; i < 8; ++i) { mrow[i] = -1e30f; lrow[i] = 0.0f; }
#pragma unroll
    for (int t = 0; t < 4; ++t) oacc[t] = vzero8();

    __shared__ __bf16 pls[16 * 16];

    for (int kt = 0; kt <= qt; ++kt) {
        // ---- scores S = Q @ K^T (K-dim = HD = 64) ----
        const __bf16* kb = Kb + ((size_t)bh * GPT_S + kt * 16) * GPT_HD;
        bf16x16 bk0 = *(const bf16x16*)(kb + ln15 * GPT_HD + hs * 16);
        bf16x16 bk1 = *(const bf16x16*)(kb + ln15 * GPT_HD + 32 + hs * 16);
        v8f s = vzero8();
        s = wmma_bf16(aq0, bk0, s);
        s = wmma_bf16(aq1, bk1, s);

        float alpha[8];
#pragma unroll
        for (int i = 0; i < 8; ++i) {
            int r = i + 8 * hs;                 // local query row of this element
            if (kt == qt && ln15 > r) s[i] = -1e30f;   // causal mask (diag tile)
            float x = s[i];
            float rm = x;
            for (int off = 1; off < 16; off <<= 1)
                rm = fmaxf(rm, __shfl_xor(rm, off, 32));
            float mn = fmaxf(mrow[i], rm);
            float al = __expf(mrow[i] - mn);
            float p  = __expf(x - mn);
            float rs = p;
            for (int off = 1; off < 16; off <<= 1)
                rs += __shfl_xor(rs, off, 32);
            lrow[i]  = lrow[i] * al + rs;
            mrow[i]  = mn;
            alpha[i] = al;
            pls[r * 16 + ln15] = (__bf16)p;     // C-layout -> LDS
        }
        __syncthreads();

        // ---- P as A-fragment (K padded 16->32 with zeros) ----
        union { bf16x16 v; bf16x8 h[2]; } ap;
        if (hs == 0) {
            ap.h[0] = *(const bf16x8*)(pls + ln15 * 16);
            ap.h[1] = *(const bf16x8*)(pls + ln15 * 16 + 8);
        } else {
#pragma unroll
            for (int e = 0; e < 8; ++e) { ap.h[0][e] = (__bf16)0.0f; ap.h[1][e] = (__bf16)0.0f; }
        }

        // ---- O = diag(alpha) * O + P @ V ----
        const __bf16* vb = Vt + (size_t)bh * GPT_HD * GPT_S;
#pragma unroll
        for (int t = 0; t < 4; ++t) {
            bf16x16 bv = *(const bf16x16*)(vb + (size_t)(t * 16 + ln15) * GPT_S + kt * 16);
            v8f cin;
#pragma unroll
            for (int i = 0; i < 8; ++i) cin[i] = oacc[t][i] * alpha[i];
            oacc[t] = wmma_bf16(ap.v, bv, cin);
        }
        __syncthreads();
    }

    // ---- normalize and write to [B,S,D] bf16 (A-matrix for proj GEMM) ----
    int b = bh >> 3;               // H = 8
    int h = bh & (GPT_H - 1);
#pragma unroll
    for (int t = 0; t < 4; ++t) {
#pragma unroll
        for (int i = 0; i < 8; ++i) {
            int srow = q0 + i + 8 * hs;
            O[((size_t)(b * GPT_S + srow)) * GPT_D + h * GPT_HD + t * 16 + ln15] =
                (__bf16)(oacc[t][i] / lrow[i]);
        }
    }
}

// ---------------------------------------------------------------------------
// Host orchestration
// ---------------------------------------------------------------------------
extern "C" void kernel_launch(void* const* d_in, const int* in_sizes, int n_in,
                              void* d_out, int out_size, void* d_ws, size_t ws_size,
                              hipStream_t stream) {
    (void)in_sizes; (void)n_in; (void)out_size; (void)ws_size;

    const int*   idx   = (const int*)d_in[0];
    const float* wte   = (const float*)d_in[1];
    const float* wpe   = (const float*)d_in[2];
    const float* ln1_w = (const float*)d_in[3];
    const float* ln1_b = (const float*)d_in[4];
    const float* Wqkv  = (const float*)d_in[5];
    const float* bqkv  = (const float*)d_in[6];
    const float* Wproj = (const float*)d_in[7];
    const float* bproj = (const float*)d_in[8];
    const float* ln2_w = (const float*)d_in[9];
    const float* ln2_b = (const float*)d_in[10];
    const float* Wfc   = (const float*)d_in[11];
    const float* bfc   = (const float*)d_in[12];
    const float* Wfc2  = (const float*)d_in[13];
    const float* bfc2  = (const float*)d_in[14];
    const float* lnf_w = (const float*)d_in[15];
    const float* lnf_b = (const float*)d_in[16];

    // ---- workspace carve-up (256B aligned) ----
    char* wsp = (char*)d_ws;
    auto alloc = [&](size_t bytes) -> void* {
        void* p = (void*)wsp;
        wsp += (bytes + 255) & ~(size_t)255;
        return p;
    };
    __bf16* wteb   = (__bf16*)alloc((size_t)GPT_V * GPT_D * 2);
    __bf16* wqkvT  = (__bf16*)alloc((size_t)GPT_L * 3 * GPT_D * GPT_D * 2);
    __bf16* wprojT = (__bf16*)alloc((size_t)GPT_L * GPT_D * GPT_D * 2);
    __bf16* wfcT   = (__bf16*)alloc((size_t)GPT_L * GPT_FF * GPT_D * 2);
    __bf16* wfc2T  = (__bf16*)alloc((size_t)GPT_L * GPT_D * GPT_FF * 2);
    float*  x      = (float*)alloc((size_t)GPT_N * GPT_D * 4);
    __bf16* hbf    = (__bf16*)alloc((size_t)GPT_N * GPT_D * 2);
    __bf16* attnbf = (__bf16*)alloc((size_t)GPT_N * GPT_D * 2);
    __bf16* ffbf   = (__bf16*)alloc((size_t)GPT_N * GPT_FF * 2);
    __bf16* Qb     = (__bf16*)alloc((size_t)GPT_N * GPT_D * 2);  // [B,H,S,HD]
    __bf16* Kb     = (__bf16*)alloc((size_t)GPT_N * GPT_D * 2);  // [B,H,S,HD]
    __bf16* Vt     = (__bf16*)alloc((size_t)GPT_N * GPT_D * 2);  // [B,H,HD,S]

    auto launch_gemm = [&](const __bf16* A, const __bf16* Bt, const float* bias,
                           float* outf, __bf16* outb, int M, int N, int K, int mode) {
        dim3 blk(32, 4);
        dim3 grd((N + 255) / 256, M / 16);
        size_t shbytes = (size_t)16 * K * 2;   // 16 x K bf16 A strip (TDM target)
        gemm_bf16_wmma<<<grd, blk, shbytes, stream>>>(A, Bt, bias, outf, outb,
                                                      Qb, Kb, Vt, M, N, K, mode);
    };
    auto launch_tr = [&](const float* in, __bf16* out, int R, int C) {
        int total = R * C;
        transpose_f32_bf16_kernel<<<(total + 255) / 256, 256, 0, stream>>>(in, out, R, C, total);
    };

    // ---- weight prep: bf16 convert / transpose (weights then live in L2) ----
    {
        int total = GPT_V * GPT_D;
        convert_f32_bf16_kernel<<<(total + 255) / 256, 256, 0, stream>>>(wte, wteb, total);
    }
    for (int l = 0; l < GPT_L; ++l) {
        launch_tr(Wqkv  + (size_t)l * GPT_D * 3 * GPT_D, wqkvT  + (size_t)l * 3 * GPT_D * GPT_D, GPT_D, 3 * GPT_D);
        launch_tr(Wproj + (size_t)l * GPT_D * GPT_D,     wprojT + (size_t)l * GPT_D * GPT_D,     GPT_D, GPT_D);
        launch_tr(Wfc   + (size_t)l * GPT_D * GPT_FF,    wfcT   + (size_t)l * GPT_FF * GPT_D,    GPT_D, GPT_FF);
        launch_tr(Wfc2  + (size_t)l * GPT_FF * GPT_D,    wfc2T  + (size_t)l * GPT_D * GPT_FF,    GPT_FF, GPT_D);
    }

    // ---- embedding ----
    {
        int total = GPT_N * GPT_D;
        embed_kernel<<<(total + 255) / 256, 256, 0, stream>>>(idx, wte, wpe, x, total);
    }

    // ---- transformer layers ----
    for (int l = 0; l < GPT_L; ++l) {
        ln_kernel<<<GPT_N, 256, 0, stream>>>(x, ln1_w + l * GPT_D, ln1_b + l * GPT_D, hbf);
        launch_gemm(hbf, wqkvT + (size_t)l * 3 * GPT_D * GPT_D, bqkv + l * 3 * GPT_D,
                    nullptr, nullptr, GPT_N, 3 * GPT_D, GPT_D, MODE_QKV);
        flash_attn_kernel<<<dim3(GPT_B * GPT_H, GPT_S / 16), 32, 0, stream>>>(Qb, Kb, Vt, attnbf);
        launch_gemm(attnbf, wprojT + (size_t)l * GPT_D * GPT_D, bproj + l * GPT_D,
                    x, nullptr, GPT_N, GPT_D, GPT_D, MODE_RESID);
        ln_kernel<<<GPT_N, 256, 0, stream>>>(x, ln2_w + l * GPT_D, ln2_b + l * GPT_D, hbf);
        launch_gemm(hbf, wfcT + (size_t)l * GPT_FF * GPT_D, bfc + l * GPT_FF,
                    nullptr, ffbf, GPT_N, GPT_FF, GPT_D, MODE_GELU);
        launch_gemm(ffbf, wfc2T + (size_t)l * GPT_D * GPT_FF, bfc2 + l * GPT_D,
                    x, nullptr, GPT_N, GPT_D, GPT_FF, MODE_RESID);
    }

    // ---- final LN + weight-tied LM head (dominant GEMM: 4096x50257x512) ----
    ln_kernel<<<GPT_N, 256, 0, stream>>>(x, lnf_w, lnf_b, hbf);
    launch_gemm(hbf, wteb, nullptr, (float*)d_out, nullptr,
                GPT_N, GPT_V, GPT_D, MODE_PLAIN);
}
